// ContrastiveEGNN_19937238188601
// MI455X (gfx1250) — compile-verified, matrix operand order checked by reading, not run on hardware
//
#include <hip/hip_runtime.h>
#include <hip/hip_bf16.h>

// ---------------------------------------------------------------------------
// ContrastiveEGNN for gfx1250 (MI455X): WMMA bf16 GEMM core + fused EGCL glue.
// Node-dimension buffers are padded to a multiple of the M-tile so every GEMM
// runs guard-free; only the external node_features input needs a clamped read.
// ---------------------------------------------------------------------------

typedef __attribute__((ext_vector_type(16))) __bf16 v16bf;
typedef __attribute__((ext_vector_type(8)))  float  v8f;

#define BM 64
#define BN 64
#define BK 32
#define LDT 40          // LDS row stride (bf16); 80B rows keep 16B-aligned chunks
#define EGNN_EPS 1e-8f

struct Frag32B { uint4 lo; uint4 hi; };   // 32 bytes == v16bf

__device__ __forceinline__ void store_bf4(__bf16* dst, float4 v) {
    union { __bf16 h[4]; uint2 u; } pk;
    pk.h[0] = (__bf16)v.x;
    pk.h[1] = (__bf16)v.y;
    pk.h[2] = (__bf16)v.z;
    pk.h[3] = (__bf16)v.w;
    *(uint2*)dst = pk.u;
}

__device__ __forceinline__ float silu_f(float x) {
    return x / (1.f + expf(-x));
}

// ---------------------------------------------------------------------------
// Generic bf16-WMMA GEMM: C[M,N] = epi(A' @ W[K,N] + bias)
//  AMODE 0: A' = A[M,K] row-major          (requires K % 32 == 0)
//  AMODE 1: A' = [h[row], h[col], radial, edge_attr]  (K = 2H+1+EA)
//  AMODE 2: A' = [h, agg]                  (K = 2H)
//  EPI 0: bias only   EPI 1: SiLU   EPI 3: +resid   EPI 4: atomic dot with wc2
//  CLAMPA: clamp A row reads to Mreal-1 (external input not padded); output
//          rows beyond Mreal are garbage written into padded buffer rows.
//  Requires M % BM == 0 (padded), N % 64 == 0, 16B-aligned rows.
//  For AMODE 1 the B k-range is clamped; valid because the A tail columns
//  beyond K are zero-filled, nulling those products.
// ---------------------------------------------------------------------------
template<int AMODE, int EPI, bool CLAMPA>
__global__ __launch_bounds__(128) void gemm_wmma_bf16(
    const float* __restrict__ A, const float* __restrict__ A2,
    const float* __restrict__ W, const float* __restrict__ bias,
    const int* __restrict__ row, const int* __restrict__ col,
    const float* __restrict__ radial, const float* __restrict__ eattr,
    const float* __restrict__ resid, const float* __restrict__ wc2,
    float* __restrict__ sdot, float* __restrict__ C,
    int M, int Mreal, int N, int K, int H, int EA)
{
    __shared__ alignas(16) __bf16 As[BM * LDT];
    __shared__ alignas(16) __bf16 Bs[BN * LDT];

    const int tid  = threadIdx.x;
    const int bm   = blockIdx.y * BM;
    const int bn   = blockIdx.x * BN;
    const int wid  = tid >> 5;
    const int lane = tid & 31;
    const int wm   = (wid >> 1) * 32;
    const int wn   = (wid & 1) * 32;

    v8f acc[2][2] = {};

    const int ksteps = (K + BK - 1) / BK;
    for (int ks = 0; ks < ksteps; ++ks) {
        const int k0 = ks * BK;

        // Branchless per-thread prefetch of the next weight tile (global_prefetch_b8).
        {
            int knext = (ks + 1 < ksteps) ? (k0 + BK) : k0;
            if (AMODE == 1) knext = min(knext, K - 1);
            __builtin_prefetch(
                &W[(long long)(knext + (tid >> 4) * 4) * N + bn + (tid & 15) * 4], 0, 1);
        }

        // ---- Stage A tile (64 x 32 f32 -> bf16): batch loads, then convert ----
        if (AMODE == 0 || AMODE == 2) {
            const float* src;
            int koff, pitch;
            if (AMODE == 0) { src = A; koff = k0; pitch = K; }
            else { src = (k0 < H) ? A : A2; koff = (k0 < H) ? k0 : (k0 - H); pitch = H; }
            float4 vv[4];
            #pragma unroll
            for (int i = 0; i < 4; ++i) {
                int idx4 = tid * 4 + i;           // 0..511
                int r    = idx4 >> 3;             // 8 float4 per row
                int c4   = (idx4 & 7) * 4;
                int gm   = bm + r;
                if (CLAMPA) gm = min(gm, Mreal - 1);   // clamp, no predication
                vv[i] = *(const float4*)&src[(long long)gm * pitch + koff + c4];
            }
            #pragma unroll
            for (int i = 0; i < 4; ++i) {
                int idx4 = tid * 4 + i;
                int r    = idx4 >> 3;
                int c4   = (idx4 & 7) * 4;
                store_bf4(&As[r * LDT + c4], vv[i]);
            }
        } else { // AMODE == 1
            const bool inRow = (k0 + BK <= H);
            const bool inCol = !inRow && (k0 + BK <= 2 * H);
            if (inRow || inCol) {
                const int* idxp = inRow ? row : col;
                const int  koff = inRow ? k0 : (k0 - H);
                float4 vv[4];
                #pragma unroll
                for (int i = 0; i < 4; ++i) {
                    int idx4 = tid * 4 + i;
                    int r    = idx4 >> 3;
                    int c4   = (idx4 & 7) * 4;
                    int src  = idxp[bm + r];      // E % BM == 0: no m guard
                    vv[i] = *(const float4*)&A[(long long)src * H + koff + c4];
                }
                #pragma unroll
                for (int i = 0; i < 4; ++i) {
                    int idx4 = tid * 4 + i;
                    int r    = idx4 >> 3;
                    int c4   = (idx4 & 7) * 4;
                    store_bf4(&As[r * LDT + c4], vv[i]);
                }
            } else {
                // tail chunk at k0 == 2H: [radial, edge_attr, zero-pad]
                // Columns with k >= K are zero-filled (required for B k-clamp).
                #pragma unroll
                for (int i = 0; i < 16; ++i) {
                    int idx = tid * 16 + i;
                    int r = idx >> 5, c = idx & 31;
                    int kk = (k0 + c) - 2 * H;
                    int e = bm + r;
                    float v = 0.f;
                    if (kk == 0)            v = radial[e];
                    else if (kk <= EA && (k0 + c) < K)
                                            v = eattr[(long long)e * EA + (kk - 1)];
                    As[r * LDT + c] = (__bf16)v;
                }
            }
        }

        // ---- Stage B tile (32 x 64, transposed into Bs[n][k]) ----
        {
            float4 vv[4];
            #pragma unroll
            for (int i = 0; i < 4; ++i) {
                int idx4 = tid * 4 + i;           // 0..511
                int kk   = idx4 >> 4;             // 16 float4 per k-row
                int n4   = (idx4 & 15) * 4;
                int kg   = k0 + kk;
                if (AMODE == 1) kg = min(kg, K - 1);   // clamp; A tail zeros cover
                vv[i] = *(const float4*)&W[(long long)kg * N + bn + n4];
            }
            #pragma unroll
            for (int i = 0; i < 4; ++i) {
                int idx4 = tid * 4 + i;
                int kk   = idx4 >> 4;
                int n4   = (idx4 & 15) * 4;
                Bs[(n4 + 0) * LDT + kk] = (__bf16)vv[i].x;
                Bs[(n4 + 1) * LDT + kk] = (__bf16)vv[i].y;
                Bs[(n4 + 2) * LDT + kk] = (__bf16)vv[i].z;
                Bs[(n4 + 3) * LDT + kk] = (__bf16)vv[i].w;
            }
        }
        __syncthreads();

        // ---- Fragments + WMMA ----
        const int half = lane >> 4;
        const int r16  = lane & 15;
        const int c0   = half ? 8 : 0;
        const int c1   = half ? 24 : 16;

        v16bf fa[2], fb[2];
        #pragma unroll
        for (int mi = 0; mi < 2; ++mi) {
            const __bf16* p = &As[(wm + mi * 16 + r16) * LDT];
            Frag32B f;
            f.lo = *(const uint4*)(p + c0);
            f.hi = *(const uint4*)(p + c1);
            fa[mi] = __builtin_bit_cast(v16bf, f);
        }
        #pragma unroll
        for (int ni = 0; ni < 2; ++ni) {
            const __bf16* p = &Bs[(wn + ni * 16 + r16) * LDT];
            Frag32B f;
            f.lo = *(const uint4*)(p + c0);
            f.hi = *(const uint4*)(p + c1);
            fb[ni] = __builtin_bit_cast(v16bf, f);
        }

        #pragma unroll
        for (int mi = 0; mi < 2; ++mi)
            #pragma unroll
            for (int ni = 0; ni < 2; ++ni)
                acc[mi][ni] = __builtin_amdgcn_wmma_f32_16x16x32_bf16(
                    false, fa[mi], false, fb[ni], (short)0, acc[mi][ni],
                    false, false);
        __syncthreads();
    }

    // ---- Epilogue (guard-free): VGPR r of lane L -> row r + 8*(L>>4), col L&15
    const int half = lane >> 4;
    const int cn   = lane & 15;

    if (EPI == 4) {
        // Fused c2 dot: accumulate both N-tiles per output row, cross-lane
        // reduce within each 16-lane half, single atomic per row.
        #pragma unroll
        for (int mi = 0; mi < 2; ++mi) {
            #pragma unroll
            for (int rr = 0; rr < 8; ++rr) {
                float t = 0.f;
                #pragma unroll
                for (int ni = 0; ni < 2; ++ni) {
                    int gn = bn + wn + ni * 16 + cn;
                    float v = acc[mi][ni][rr] + bias[gn];
                    t += silu_f(v) * wc2[gn];
                }
                #pragma unroll
                for (int off = 1; off < 16; off <<= 1) t += __shfl_xor(t, off, 32);
                int gm = bm + wm + mi * 16 + rr + half * 8;
                if (cn == 0) atomicAdd(&sdot[gm], t);
            }
        }
    } else {
        #pragma unroll
        for (int mi = 0; mi < 2; ++mi) {
            #pragma unroll
            for (int ni = 0; ni < 2; ++ni) {
                int gn = bn + wn + ni * 16 + cn;
                float b = bias[gn];
                long long base =
                    (long long)(bm + wm + mi * 16 + half * 8) * N + gn;
                #pragma unroll
                for (int rr = 0; rr < 8; ++rr) {
                    float o = acc[mi][ni][rr] + b;
                    if (EPI == 1) o = silu_f(o);
                    if (EPI == 3) o += resid[base + (long long)rr * N];
                    C[base + (long long)rr * N] = o;
                }
            }
        }
    }
}

// ---------------------------------------------------------------------------
// Support kernels
// ---------------------------------------------------------------------------
__global__ void k_zero(float* __restrict__ p, long long n) {
    long long i = (long long)blockIdx.x * blockDim.x + threadIdx.x;
    if (i < n) p[i] = 0.f;
}

__global__ void k_copy(float* __restrict__ d, const float* __restrict__ s, long long n) {
    long long i = (long long)blockIdx.x * blockDim.x + threadIdx.x;
    if (i < n) d[i] = s[i];
}

__global__ void k_count_rows(const int* __restrict__ row, float* __restrict__ cnt, int E) {
    int e = blockIdx.x * blockDim.x + threadIdx.x;
    if (e < E) atomicAdd(&cnt[row[e]], 1.f);
}

__global__ void k_edge_pre(const float* __restrict__ x,
                           const int* __restrict__ row, const int* __restrict__ col,
                           float* __restrict__ dbuf, float* __restrict__ rbuf, int E) {
    int e = blockIdx.x * blockDim.x + threadIdx.x;
    if (e >= E) return;
    int r = row[e], c = col[e];
    float dx = x[r * 3 + 0] - x[c * 3 + 0];
    float dy = x[r * 3 + 1] - x[c * 3 + 1];
    float dz = x[r * 3 + 2] - x[c * 3 + 2];
    float rad = dx * dx + dy * dy + dz * dz;
    float inv = 1.f / (sqrtf(rad) + EGNN_EPS);
    dbuf[e * 3 + 0] = dx * inv;
    dbuf[e * 3 + 1] = dy * inv;
    dbuf[e * 3 + 2] = dz * inv;
    rbuf[e] = rad;
}

// One wave (32 lanes) per edge: gate = sigmoid(m . w_att + b); m *= gate.
__global__ void k_gate_scale(float* __restrict__ m,
                             const float* __restrict__ attw, const float* __restrict__ attb,
                             int E, int H) {
    int wid = blockIdx.x * (blockDim.x >> 5) + (threadIdx.x >> 5);
    if (wid >= E) return;
    int lane = threadIdx.x & 31;
    float* mr = m + (long long)wid * H;
    float s = 0.f;
    for (int k = lane; k < H; k += 32) s += mr[k] * attw[k];
    #pragma unroll
    for (int off = 16; off > 0; off >>= 1) s += __shfl_xor(s, off, 32);
    float gate = 1.f / (1.f + expf(-(s + attb[0])));
    for (int k = lane; k < H; k += 32) mr[k] *= gate;
}

// Segment-sum via global float atomics: one block per edge, blockDim == H.
__global__ void k_aggregate(const float* __restrict__ m, const float* __restrict__ dbuf,
                            const float* __restrict__ sdot, const int* __restrict__ row,
                            float* __restrict__ aggm, float* __restrict__ aggc,
                            int E, int H) {
    int e = blockIdx.x;
    if (e >= E) return;
    int t = threadIdx.x;
    int r = row[e];
    atomicAdd(&aggm[(long long)r * H + t], m[(long long)e * H + t]);
    if (t < 3) atomicAdd(&aggc[r * 3 + t], dbuf[e * 3 + t] * sdot[e]);
}

__global__ void k_x_update(float* __restrict__ x, const float* __restrict__ aggc,
                           const float* __restrict__ cnt, int n) {
    int i = blockIdx.x * blockDim.x + threadIdx.x;
    if (i >= n) return;
    float c = fmaxf(cnt[i], 1.f);
    x[i * 3 + 0] += aggc[i * 3 + 0] / c;
    x[i * 3 + 1] += aggc[i * 3 + 1] / c;
    x[i * 3 + 2] += aggc[i * 3 + 2] / c;
}

// One block per feature column: g[f] = mean over nodes.
__global__ void k_mean_pool(const float* __restrict__ h, float* __restrict__ g,
                            int n, int H) {
    __shared__ float red[256];
    int f = blockIdx.x;
    float s = 0.f;
    for (int i = threadIdx.x; i < n; i += blockDim.x) s += h[(long long)i * H + f];
    red[threadIdx.x] = s;
    __syncthreads();
    for (int o = 128; o > 0; o >>= 1) {
        if ((int)threadIdx.x < o) red[threadIdx.x] += red[threadIdx.x + o];
        __syncthreads();
    }
    if (threadIdx.x == 0) g[f] = red[0] / (float)n;
}

// out[O] = (relu(g @ W1 + b1)) @ W2 + b2, single block of 128 threads.
__global__ void k_fe_head(const float* __restrict__ g,
                          const float* __restrict__ w1, const float* __restrict__ b1,
                          const float* __restrict__ w2, const float* __restrict__ b2,
                          float* __restrict__ out, int H, int P, int O) {
    __shared__ float p[128];
    int t = threadIdx.x;
    if (t < P) {
        float s = b1[t];
        for (int k = 0; k < H; ++k) s += g[k] * w1[k * P + t];
        p[t] = fmaxf(s, 0.f);
    }
    __syncthreads();
    if (t < O) {
        float s = b2[t];
        for (int k = 0; k < P; ++k) s += p[k] * w2[k * O + t];
        out[t] = s;
    }
}

// ---------------------------------------------------------------------------
// Host orchestration
// ---------------------------------------------------------------------------
struct GemmArgs {
    const float *A, *A2, *W, *b, *resid, *wc2;
    float *sd, *C;
    const int *row, *col;
    const float *radial, *eattr;
    int M, Mreal, N, K, H, EA;
};

template<int AMODE, int EPI, bool CLAMPA>
static void launch_one(const GemmArgs& a, hipStream_t stream) {
    dim3 grid((a.N + BN - 1) / BN, (a.M + BM - 1) / BM);
    gemm_wmma_bf16<AMODE, EPI, CLAMPA><<<grid, 128, 0, stream>>>(
        a.A, a.A2, a.W, a.b, a.row, a.col, a.radial, a.eattr,
        a.resid, a.wc2, a.sd, a.C, a.M, a.Mreal, a.N, a.K, a.H, a.EA);
}

// sel: 0 emb (plain, nodes, clamp-A), 1 e1 (edge concat), 2 e2 (edges, silu),
//      3 c1 fused-dot (edges), 4 n1 (node concat, silu), 5 n2 (resid, nodes)
static void launch_gemm(int sel, const GemmArgs& a, hipStream_t stream) {
    switch (sel) {
        case 0: launch_one<0, 0, true >(a, stream); break;
        case 1: launch_one<1, 1, false>(a, stream); break;
        case 2: launch_one<0, 1, false>(a, stream); break;
        case 3: launch_one<0, 4, false>(a, stream); break;
        case 4: launch_one<2, 1, false>(a, stream); break;
        case 5: launch_one<0, 3, false>(a, stream); break;
    }
}

extern "C" void kernel_launch(void* const* d_in, const int* in_sizes, int n_in,
                              void* d_out, int out_size, void* d_ws, size_t ws_size,
                              hipStream_t stream) {
    (void)in_sizes; (void)n_in; (void)out_size; (void)ws_size;

    const int NN = 6000, NP = 6016;   // NP: node rows padded to multiple of BM
    const int E = 96000, EA = 11;
    const int HID1 = 512, HID2 = 128, IN_NODE = 960;

    const float* nodef = (const float*)d_in[0];
    const int*   eidx  = (const int*)d_in[1];
    const float* pos   = (const float*)d_in[2];
    const float* eattr = (const float*)d_in[3];
    const void* const* P = d_in + 4;   // 120 param leaves (pytree, dict keys sorted)
    auto Pf = [&](int i) { return (const float*)P[i]; };

    const int* drow = eidx;
    const int* dcol = eidx + E;

    // Workspace carve (node buffers padded to NP rows)
    char* wsb = (char*)d_ws;
    size_t off = 0;
    auto carve = [&](size_t bytes) {
        void* p = wsb + off;
        off += (bytes + 255) & ~(size_t)255;
        return p;
    };
    float* hA   = (float*)carve((size_t)NP * HID1 * 4);
    float* hB   = (float*)carve((size_t)NP * HID1 * 4);
    float* mA   = (float*)carve((size_t)E * HID1 * 4);
    float* mB   = (float*)carve((size_t)E * HID1 * 4);
    float* aggm = (float*)carve((size_t)NP * HID1 * 4);
    float* dbuf = (float*)carve((size_t)E * 3 * 4);
    float* rbuf = (float*)carve((size_t)E * 4);
    float* sdot = (float*)carve((size_t)E * 4);
    float* xbuf = (float*)carve((size_t)NN * 3 * 4);
    float* aggc = (float*)carve((size_t)NN * 3 * 4);
    float* cnt  = (float*)carve((size_t)NN * 4);
    float* gvec = (float*)carve(256 * 4);

    auto zero = [&](float* p, long long n) {
        k_zero<<<(unsigned)((n + 255) / 256), 256, 0, stream>>>(p, n);
    };
    auto gemm = [&](int sel, const float* A, const float* A2,
                    const float* W, const float* b, const float* resid,
                    const float* wc2, float* sd, float* C,
                    int M, int Mreal, int N, int K, int H) {
        GemmArgs a;
        a.A = A; a.A2 = A2; a.W = W; a.b = b; a.resid = resid; a.wc2 = wc2;
        a.sd = sd; a.C = C; a.row = drow; a.col = dcol;
        a.radial = rbuf; a.eattr = eattr;
        a.M = M; a.Mreal = Mreal; a.N = N; a.K = K; a.H = H; a.EA = EA;
        launch_gemm(sel, a, stream);
    };

    // Static per-launch setup (deterministic; redone every call).
    zero(cnt, NN);
    k_count_rows<<<(E + 255) / 256, 256, 0, stream>>>(drow, cnt, E);
    k_copy<<<(NN * 3 + 255) / 256, 256, 0, stream>>>(xbuf, pos, (long long)NN * 3);

    float* cur = hA;
    float* nxt = hB;

    for (int gi = 0; gi < 2; ++gi) {
        const int base = gi * 56;
        const int H    = gi ? HID2 : HID1;
        const int Kin  = gi ? HID1 : IN_NODE;
        const float* hin = gi ? cur : nodef;
        const int Mclamp = gi ? NP : NN;   // gi=0 reads unpadded external input

        // emb_in (Linear, no activation); grid over padded NP rows
        gemm(0, hin, nullptr, Pf(base + 0), Pf(base + 1),
             nullptr, nullptr, nullptr, nxt, NP, Mclamp, H, Kin, H);
        { float* t = cur; cur = nxt; nxt = t; }

        for (int l = 0; l < 4; ++l) {
            const int L = base + 4 + 13 * l;
            const float* attw = Pf(L + 0);
            const float* attb = Pf(L + 1);
            const float* c1w  = Pf(L + 2);
            const float* c1b  = Pf(L + 3);
            const float* c2w  = Pf(L + 4);
            const float* e1w  = Pf(L + 5);
            const float* e1b  = Pf(L + 6);
            const float* e2w  = Pf(L + 7);
            const float* e2b  = Pf(L + 8);
            const float* n1w  = Pf(L + 9);
            const float* n1b  = Pf(L + 10);
            const float* n2w  = Pf(L + 11);
            const float* n2b  = Pf(L + 12);

            k_edge_pre<<<(E + 255) / 256, 256, 0, stream>>>(xbuf, drow, dcol, dbuf, rbuf, E);

            // edge MLP: m1 = silu(e1(concat)), m = silu(e2(m1))
            gemm(1, cur, nullptr, e1w, e1b, nullptr, nullptr, nullptr,
                 mA, E, E, H, 2 * H + 1 + EA, H);
            gemm(2, mA, nullptr, e2w, e2b, nullptr, nullptr, nullptr,
                 mB, E, E, H, H, H);
            k_gate_scale<<<(E + 7) / 8, 256, 0, stream>>>(mB, attw, attb, E, H);

            // coord path: s = silu(c1(m)) . c2 (fused dot epilogue)
            zero(sdot, E);
            zero(aggm, (long long)NP * H);
            zero(aggc, (long long)NN * 3);
            gemm(3, mB, nullptr, c1w, c1b, nullptr, c2w, sdot,
                 nullptr, E, E, H, H, H);

            k_aggregate<<<E, H, 0, stream>>>(mB, dbuf, sdot, drow, aggm, aggc, E, H);
            k_x_update<<<(NN + 255) / 256, 256, 0, stream>>>(xbuf, aggc, cnt, NN);

            // node MLP with residual: h = h + n2(silu(n1([h, agg])))
            gemm(4, cur, aggm, n1w, n1b, nullptr, nullptr, nullptr,
                 mA /* t reuse */, NP, NP, H, 2 * H, H);
            gemm(5, mA, nullptr, n2w, n2b, cur, nullptr, nullptr,
                 nxt, NP, NP, H, H, H);
            { float* t = cur; cur = nxt; nxt = t; }
        }

        // emb_out (Linear, no activation)
        gemm(0, cur, nullptr, Pf(base + 2), Pf(base + 3),
             nullptr, nullptr, nullptr, nxt, NP, NP, H, H, H);
        { float* t = cur; cur = nxt; nxt = t; }
    }

    // mean pool over nodes -> g[128] (only real rows)
    k_mean_pool<<<HID2, 256, 0, stream>>>(cur, gvec, NN, HID2);

    // heads: fe1 -> d_out[0:64], fe2 -> d_out[64:128]
    float* outp = (float*)d_out;
    k_fe_head<<<1, 128, 0, stream>>>(gvec, Pf(112), Pf(113), Pf(114), Pf(115),
                                     outp, HID2, 96, 64);
    k_fe_head<<<1, 128, 0, stream>>>(gvec, Pf(116), Pf(117), Pf(118), Pf(119),
                                     outp + 64, HID2, 96, 64);
}
